// WaveNet_10599979286805
// MI455X (gfx1250) — compile-verified, matrix operand order checked by reading, not run on hardware
//
#include <hip/hip_runtime.h>

typedef _Float16 half_t;
typedef __attribute__((ext_vector_type(16))) _Float16 v16h;
typedef __attribute__((ext_vector_type(8)))  _Float16 v8h;
typedef __attribute__((ext_vector_type(8)))  float    v8f;

// ---- d_ws / LDS weight image layout (offsets in halves) ----
#define W1T_OFF 0              // [64][40]   K=12 padded->32, row stride 40
#define W1T_S   40
#define W2T_OFF 2560           // [128][72]  K=64, stride 72
#define W2T_S   72
#define W3T_OFF 11776          // [256][136] K=128, stride 136
#define W3T_S   136
#define W4T_OFF 46592          // [128][264] K=256, stride 264
#define W4T_S   264
#define W5T_OFF 80384          // [64][136]  K=128, stride 136
#define W5T_S   136
#define WH_TOTAL 89088
#define WBYTES  (WH_TOTAL * 2)          // 178176, 16B aligned

// float param block (indices into float region)
#define FB_B1   0
#define FB_B2   64
#define FB_B3   192
#define FB_B4   448
#define FB_B5   576
#define FB_W6   640
#define FB_B6   704
#define FB_WT1  705
#define FB_BT1  721
#define FB_WT2  737
#define FB_BT2  753
#define NFLOATS 754

#define ACT_BASE 181200                 // 16B aligned LDS offset for activations
#define AS       264                    // activation row stride in halves
#define ACT_PER_WAVE (2 * 16 * AS)      // ping+pong, halves
#define SMEM_BYTES (ACT_BASE + 8 * ACT_PER_WAVE * 2)   // 316368 B <= 320KB WGP LDS

// hw v_sin/v_cos compute sin/cos(2*pi*x): sin(pi*v) == hwsin(0.5*v)
#define HSIN(x) __builtin_amdgcn_sinf(x)
#define HCOS(x) __builtin_amdgcn_cosf(x)

// ------------------------------------------------------------------
// Prep: f32 [in][out] weights -> f16 transposed/padded image in d_ws
// ------------------------------------------------------------------
__global__ void prep_weights(const float* __restrict__ W1, const float* __restrict__ W2,
                             const float* __restrict__ W3, const float* __restrict__ W4,
                             const float* __restrict__ W5,
                             const float* __restrict__ b1, const float* __restrict__ b2,
                             const float* __restrict__ b3, const float* __restrict__ b4,
                             const float* __restrict__ b5,
                             const float* __restrict__ W6, const float* __restrict__ b6,
                             const float* __restrict__ Wt1, const float* __restrict__ bt1,
                             const float* __restrict__ Wt2, const float* __restrict__ bt2,
                             half_t* __restrict__ wh, float* __restrict__ wf)
{
    int i = blockIdx.x * blockDim.x + threadIdx.x;
    if (i < WH_TOTAL) {
        float v = 0.0f;
        if (i < W2T_OFF)      { int j = i - W1T_OFF; int o = j / W1T_S, k = j % W1T_S; if (k < 12)  v = W1[k * 64  + o]; }
        else if (i < W3T_OFF) { int j = i - W2T_OFF; int o = j / W2T_S, k = j % W2T_S; if (k < 64)  v = W2[k * 128 + o]; }
        else if (i < W4T_OFF) { int j = i - W3T_OFF; int o = j / W3T_S, k = j % W3T_S; if (k < 128) v = W3[k * 256 + o]; }
        else if (i < W5T_OFF) { int j = i - W4T_OFF; int o = j / W4T_S, k = j % W4T_S; if (k < 256) v = W4[k * 128 + o]; }
        else                  { int j = i - W5T_OFF; int o = j / W5T_S, k = j % W5T_S; if (k < 128) v = W5[k * 64  + o]; }
        wh[i] = (half_t)v;
    } else if (i < WH_TOTAL + NFLOATS) {
        int j = i - WH_TOTAL;
        float v;
        if (j < FB_B2)      v = b1[j - FB_B1];
        else if (j < FB_B3) v = b2[j - FB_B2];
        else if (j < FB_B4) v = b3[j - FB_B3];
        else if (j < FB_B5) v = b4[j - FB_B4];
        else if (j < FB_W6) v = b5[j - FB_B5];
        else if (j < FB_B6) v = W6[j - FB_W6];
        else if (j < FB_WT1) v = b6[0];
        else if (j < FB_BT1) v = Wt1[j - FB_WT1];
        else if (j < FB_WT2) v = bt1[j - FB_BT1];
        else if (j < FB_BT2) v = Wt2[j - FB_WT2];
        else                 v = bt2[0];
        wf[j] = v;
    }
}

// ------------------------------------------------------------------
// WMMA helpers
// ------------------------------------------------------------------
// 16x32 f16 A/B fragment from LDS: lane<16 -> row=lane, K=[k0..k0+7]+[k0+16..k0+23]
//                                  lane>=16 -> row=lane-16, K shifted by +8 (ISA 7.12.2)
__device__ __forceinline__ v16h load_frag(const half_t* base, int stride, int k0, int lane)
{
    const int row = lane & 15;
    const int kb  = k0 + ((lane >> 4) << 3);
    union { v16h v; v8h h[2]; } u;
    u.h[0] = *(const v8h*)(base + row * stride + kb);
    u.h[1] = *(const v8h*)(base + row * stride + kb + 16);
    return u.v;
}

// Dense layer: D = sin(pi*(A@W + b)) * (sign folded into scale), A tile kept in VGPRs.
template<int KP, int NOUT, int WS>
__device__ __forceinline__ void dense_sin(const half_t* __restrict__ src,
                                          half_t* __restrict__ dst,
                                          const half_t* __restrict__ wt,
                                          const float* __restrict__ bias,
                                          float scale,   // +0.5 or -0.5
                                          int lane)
{
    const int col = lane & 15;
    const int hi  = (lane >> 4) << 3;
    constexpr int KF = KP / 32;

    // hoist all A fragments for this layer into registers (<= 8 x 8 VGPRs)
    v16h afr[KF];
#pragma unroll
    for (int k = 0; k < KF; ++k) afr[k] = load_frag(src, AS, k * 32, lane);

    for (int n0 = 0; n0 < NOUT; n0 += 16) {
        const float bn = bias[n0 + col];
        v8f acc;
#pragma unroll
        for (int r = 0; r < 8; ++r) acc[r] = bn;    // bias rides the C operand
#pragma unroll
        for (int k = 0; k < KF; ++k) {
            v16h b = load_frag(wt + n0 * WS, WS, k * 32, lane);
            acc = __builtin_amdgcn_wmma_f32_16x16x32_f16(
                      false, afr[k], false, b, (short)0, acc, false, false);
        }
#pragma unroll
        for (int r = 0; r < 8; ++r) {
            // sgn*sin(pi*v) == hwsin(sgn*0.5*v)
            dst[(r + hi) * AS + n0 + col] = (half_t)HSIN(scale * acc[r]);
        }
    }
    __builtin_amdgcn_wave_barrier();
}

// ------------------------------------------------------------------
// Main fused MLP kernel: each wave owns a 16-point tile end-to-end
// ------------------------------------------------------------------
__global__ void __launch_bounds__(256)
pinn_mlp(const float* __restrict__ xyt, const half_t* __restrict__ wh,
         const float* __restrict__ wf, float* __restrict__ out, int N)
{
    extern __shared__ char smem[];
    float* swf = (float*)(smem + WBYTES);

    // one-time cooperative weight image load: global -> LDS
    {
        const uint4* g = (const uint4*)wh;
        uint4*       s = (uint4*)smem;
        for (int i = threadIdx.x; i < WBYTES / 16; i += blockDim.x) s[i] = g[i];
        for (int i = threadIdx.x; i < NFLOATS;     i += blockDim.x) swf[i] = wf[i];
    }
    __syncthreads();

    const half_t* swh = (const half_t*)smem;
    const int lane = threadIdx.x & 31;
    const int wave = threadIdx.x >> 5;
    half_t* buf0 = (half_t*)(smem + ACT_BASE) + wave * ACT_PER_WAVE;
    half_t* buf1 = buf0 + 16 * AS;

    const int tiles  = (N + 15) >> 4;
    const int nwaves = gridDim.x * (blockDim.x >> 5);

    for (int tile = blockIdx.x * (blockDim.x >> 5) + wave; tile < tiles; tile += nwaves) {
        const int r0 = tile << 4;
        float tval = 0.0f;

        // Fourier embed x,y into buf0[16][0..11], zero-pad K to 32; time branch per row.
        {
            const int row = lane & 15;
            half_t* a0 = buf0 + row * AS;
            if (lane < 16) {
                const int r = r0 + row;
                float x = 0.f, y = 0.f, t = 0.f;
                if (r < N) { x = xyt[3 * r]; y = xyt[3 * r + 1]; t = xyt[3 * r + 2]; }
                // sin(pi*f*x) == hwsin(0.5*f*x), f = 1,2,4
                a0[0]  = (half_t)HSIN(0.5f * x);
                a0[1]  = (half_t)HSIN(x);
                a0[2]  = (half_t)HSIN(2.0f * x);
                a0[3]  = (half_t)HCOS(0.5f * x);
                a0[4]  = (half_t)HCOS(x);
                a0[5]  = (half_t)HCOS(2.0f * x);
                a0[6]  = (half_t)HSIN(0.5f * y);
                a0[7]  = (half_t)HSIN(y);
                a0[8]  = (half_t)HSIN(2.0f * y);
                a0[9]  = (half_t)HCOS(0.5f * y);
                a0[10] = (half_t)HCOS(y);
                a0[11] = (half_t)HCOS(2.0f * y);
                tval = swf[FB_BT2];
#pragma unroll
                for (int j = 0; j < 16; ++j)
                    tval += swf[FB_WT2 + j] * HSIN(0.5f * (t * swf[FB_WT1 + j] + swf[FB_BT1 + j]));
            } else {
#pragma unroll
                for (int k = 12; k < 32; ++k) a0[k] = (half_t)0.0f;
            }
        }
        __builtin_amdgcn_wave_barrier();

        dense_sin< 32,  64, W1T_S>(buf0, buf1, swh + W1T_OFF, swf + FB_B1, +0.5f, lane);
        dense_sin< 64, 128, W2T_S>(buf1, buf0, swh + W2T_OFF, swf + FB_B2, -0.5f, lane);
        dense_sin<128, 256, W3T_S>(buf0, buf1, swh + W3T_OFF, swf + FB_B3, +0.5f, lane);
        dense_sin<256, 128, W4T_S>(buf1, buf0, swh + W4T_OFF, swf + FB_B4, -0.5f, lane);
        dense_sin<128,  64, W5T_S>(buf0, buf1, swh + W5T_OFF, swf + FB_B5, +0.5f, lane);

        // Final 64 -> 1 (linear) * time branch
        if (lane < 16) {
            const int r = r0 + lane;
            if (r < N) {
                float s = swf[FB_B6];
                const half_t* a = buf1 + lane * AS;
#pragma unroll
                for (int k = 0; k < 64; ++k) s += (float)a[k] * swf[FB_W6 + k];
                out[r] = tval * s;
            }
        }
        __builtin_amdgcn_wave_barrier();
    }
}

// ------------------------------------------------------------------
extern "C" void kernel_launch(void* const* d_in, const int* in_sizes, int n_in,
                              void* d_out, int out_size, void* d_ws, size_t ws_size,
                              hipStream_t stream)
{
    const float* xyt = (const float*)d_in[0];
    const float* W1  = (const float*)d_in[1];
    const float* b1  = (const float*)d_in[2];
    const float* W2  = (const float*)d_in[3];
    const float* b2  = (const float*)d_in[4];
    const float* W3  = (const float*)d_in[5];
    const float* b3  = (const float*)d_in[6];
    const float* W4  = (const float*)d_in[7];
    const float* b4  = (const float*)d_in[8];
    const float* W5  = (const float*)d_in[9];
    const float* b5  = (const float*)d_in[10];
    const float* W6  = (const float*)d_in[11];
    const float* b6  = (const float*)d_in[12];
    const float* Wt1 = (const float*)d_in[13];
    const float* bt1 = (const float*)d_in[14];
    const float* Wt2 = (const float*)d_in[15];
    const float* bt2 = (const float*)d_in[16];

    half_t* wh = (half_t*)d_ws;
    float*  wf = (float*)((char*)d_ws + WBYTES);

    const int N = in_sizes[0] / 3;

    const int prep_total = WH_TOTAL + NFLOATS;
    prep_weights<<<(prep_total + 255) / 256, 256, 0, stream>>>(
        W1, W2, W3, W4, W5, b1, b2, b3, b4, b5, W6, b6, Wt1, bt1, Wt2, bt2, wh, wf);

    pinn_mlp<<<1024, 256, SMEM_BYTES, stream>>>(xyt, wh, wf, (float*)d_out, N);
}
